// RATEncoder_38663295599331
// MI455X (gfx1250) — compile-verified, matrix operand order checked by reading, not run on hardware
//
#include <hip/hip_runtime.h>
#include <hip/hip_bf16.h>
#include <math.h>

// ---------------- shapes (fixed by the reference) ----------------
#define BATCH   4
#define SEQ     512
#define HIDDEN  256
#define NHEAD   8
#define DHEAD   32
#define FFDIM   1024
#define RVOC    51
#define RVP     64          // padded rel vocab
#define NLAYER  4
#define MROWS   (BATCH*SEQ) // 2048

typedef __attribute__((ext_vector_type(16))) _Float16 v16h;
typedef __attribute__((ext_vector_type(8)))  float    v8f;
typedef __attribute__((ext_vector_type(4)))  int      v4i;

// ---------------- CDNA5 async global->LDS path (guarded) ----------------
#if __has_builtin(__builtin_amdgcn_global_load_async_to_lds_b32) && \
    __has_builtin(__builtin_amdgcn_global_load_async_to_lds_b128)
#define HAVE_ASYNC_LDS 1
#else
#define HAVE_ASYNC_LDS 0
#endif

#if HAVE_ASYNC_LDS
// Builtin prototypes (from hipcc diagnostics):
//   b32 : (int  AS1*, int  AS3*, imm offset, imm cpol)
//   b128: (v4i  AS1*, v4i  AS3*, imm offset, imm cpol)
typedef __attribute__((address_space(1))) int  g_i32;
typedef __attribute__((address_space(3))) int  l_i32;
typedef __attribute__((address_space(1))) v4i  g_v4i;
typedef __attribute__((address_space(3))) v4i  l_v4i;

__device__ __forceinline__ void async_b128(const void* g, void* l) {
  __builtin_amdgcn_global_load_async_to_lds_b128((g_v4i*)g, (l_v4i*)l, 0, 0);
}
template <int N>
__device__ __forceinline__ void wait_async() {
#if __has_builtin(__builtin_amdgcn_s_wait_asynccnt)
  __builtin_amdgcn_s_wait_asynccnt(N);
#else
  asm volatile("s_wait_asynccnt %0" ::"i"(N) : "memory");
#endif
}
#endif

// ---------------- WMMA helpers (gfx1250, wave32) ----------------
__device__ __forceinline__ v8f wmma_f16(v16h a, v16h b, v8f c) {
  // D = A(16x32 f16) x B(32x16 f16) + C(16x16 f32)
  return __builtin_amdgcn_wmma_f32_16x16x32_f16(
      /*neg_a*/false, a, /*neg_b*/false, b,
      /*c_mod*/(short)0, c, /*reuse_a*/false, /*reuse_b*/false);
}

// A-fragment loader from a row-major f16 LDS tile.
// Lane L: row = L%16 (passed in), half = L/16. Element j (=2v+s):
//   k = (v/4)*16 + half*8 + (v%4)*2 + s      (ISA 7.12.2, 16-bit A 16x32)
// For the B operand the SAME pattern applies if the tile is stored N-major
// (i.e. tile[n][k] == B[k][n]).
__device__ __forceinline__ v16h load_frag_h(const _Float16* tile, int row,
                                            int half, int ld) {
  const _Float16* p = tile + row * ld;
  v16h f;
#pragma unroll
  for (int v = 0; v < 8; ++v) {
    int kb = (v >> 2) * 16 + half * 8 + (v & 3) * 2;
    f[2 * v]     = p[kb];
    f[2 * v + 1] = p[kb + 1];
  }
  return f;
}

// Same pattern but reading fp32 LDS and converting to f16 on the fly.
__device__ __forceinline__ v16h load_frag_f32(const float* tile, int row,
                                              int half, int ld) {
  const float* p = tile + row * ld;
  v16h f;
#pragma unroll
  for (int v = 0; v < 8; ++v) {
    int kb = (v >> 2) * 16 + half * 8 + (v & 3) * 2;
    f[2 * v]     = (_Float16)p[kb];
    f[2 * v + 1] = (_Float16)p[kb + 1];
  }
  return f;
}

// ---------------- weight pre-pass: fp32 [K][N] -> f16 N-major [N][K] ----------------
__global__ __launch_bounds__(256) void cvt_transpose_kernel(
    const float* __restrict__ in, _Float16* __restrict__ out, int K, int N) {
  int idx = blockIdx.x * 256 + threadIdx.x;
  if (idx >= K * N) return;
  int n = idx / K, k = idx - n * K;
  out[idx] = (_Float16)in[(size_t)k * N + n];
}

// ---------------- generic GEMM: C = act(A[MxK] @ Wh[NxK,f16] + bias) ---------------
// block = 128 threads (4 waves), tile 64x64, K stepped by 32.
// Double-buffered via GLOBAL_LOAD_ASYNC_TO_LDS (ASYNCcnt): per stage each wave
// issues 4 b128 (A tile, fp32) + 2 b128 (W tile, pre-converted f16, contiguous
// on both sides) = 6 async ops; async loads retire in order, so
// s_wait_asynccnt<=6 after issuing the next batch means the previous batch is
// resident.  Only the single A fragment converts f32->f16 in the hot loop.
#define GEMM_ASYNC_OPS 6

__global__ __launch_bounds__(128) void gemm_kernel(
    const float* __restrict__ A, const _Float16* __restrict__ Wh,
    const float* __restrict__ bias, float* __restrict__ C,
    int M, int K, int N, int do_relu) {
  __shared__ float    Atf[2][64][36];  // fp32, 144B rows (16B aligned)
  __shared__ _Float16 Wt[2][64][40];   // f16 N-major, 80B rows (16B aligned)

  const int tid  = threadIdx.x;
  const int lane = tid & 31;
  const int wave = tid >> 5;
  const int m0 = blockIdx.y * 64;
  const int n0 = blockIdx.x * 64;
  const int m_lane = lane & 15;
  const int half   = lane >> 4;

  v8f acc[4] = {};

  auto stage = [&](int buf, int k0) {
#if HAVE_ASYNC_LDS
#pragma unroll
    for (int j = 0; j < 4; ++j) {  // A tile: 512 x b128 (4 consecutive floats)
      int idx = j * 128 + tid;
      int r = idx >> 3, c4 = idx & 7;
      async_b128(&A[(size_t)(m0 + r) * K + k0 + c4 * 4], &Atf[buf][r][c4 * 4]);
    }
#pragma unroll
    for (int j = 0; j < 2; ++j) {  // W tile: 256 x b128 (8 consecutive f16)
      int idx = j * 128 + tid;
      int n = idx >> 2, c8 = idx & 3;
      async_b128(&Wh[(size_t)(n0 + n) * K + k0 + c8 * 8], &Wt[buf][n][c8 * 8]);
    }
#else
    for (int i = tid; i < 64 * 32; i += 128) {
      int r = i >> 5, c = i & 31;
      Atf[buf][r][c] = A[(size_t)(m0 + r) * K + k0 + c];
    }
    for (int i = tid; i < 64 * 32; i += 128) {
      int n = i >> 5, c = i & 31;
      Wt[buf][n][c] = Wh[(size_t)(n0 + n) * K + k0 + c];
    }
#endif
  };

  stage(0, 0);
  const int nsteps = K >> 5;
  for (int s = 0; s < nsteps; ++s) {
    const int buf = s & 1;
    const bool has_next = (s + 1 < nsteps);
    if (has_next) stage(buf ^ 1, (s + 1) << 5);
#if HAVE_ASYNC_LDS
    if (has_next) wait_async<GEMM_ASYNC_OPS>();
    else          wait_async<0>();
#endif
    __syncthreads();

    v16h af = load_frag_f32(&Atf[buf][wave * 16][0], m_lane, half, 36);
#pragma unroll
    for (int nt = 0; nt < 4; ++nt) {
      v16h bf = load_frag_h(&Wt[buf][nt * 16][0], m_lane, half, 40);
      acc[nt] = wmma_f16(af, bf, acc[nt]);
    }
    __syncthreads();
  }

#pragma unroll
  for (int nt = 0; nt < 4; ++nt) {
#pragma unroll
    for (int r = 0; r < 8; ++r) {
      int m = m0 + wave * 16 + r + 8 * half;
      int n = n0 + nt * 16 + m_lane;
      float val = acc[nt][r] + (bias ? bias[n] : 0.0f);
      if (do_relu) val = fmaxf(val, 0.0f);
      C[(size_t)m * N + n] = val;
    }
  }
}

// ---------------- fused relative attention ----------------
// One wave per (b, h, 16-row q tile).  grid = B*NH*(SEQ/16) = 1024.
// Rel-position terms use the 51-entry-vocab factorization:
//   scores_rel = gather(qrel = q @ relK^T), ctx_rel = bincount(probs) @ relV
// which avoids materializing the (B,S,S,DH) gather tensors entirely.
__global__ __launch_bounds__(32) void attn_kernel(
    const float* __restrict__ qg, const float* __restrict__ kg,
    const float* __restrict__ vg, const float* __restrict__ relK,
    const float* __restrict__ relV, const int* __restrict__ rel,
    const unsigned char* __restrict__ pmask, float* __restrict__ ctx) {
  __shared__ float         scores[16][SEQ];       // 32 KB
  __shared__ unsigned char relid[16][SEQ];        //  8 KB
  __shared__ _Float16      qtile[16][40];
  __shared__ _Float16      relk_h[RVP][40];       // vocab-major == B layout for q@relK^T
  __shared__ _Float16      ktile[16][40];
  __shared__ _Float16      vtile[32][40];         // d-major (transposed V chunk)
  __shared__ _Float16      relvt[32][72];         // relvt[d][r] = relV[r][d]
  __shared__ float         qrel[16][RVP];
  __shared__ float         prbin[16][RVP];

  const int lane = threadIdx.x;
  const int bid  = blockIdx.x;
  const int qt = bid & 31;
  const int h  = (bid >> 5) & 7;
  const int b  = bid >> 8;
  const int row0 = b * SEQ + qt * 16;
  const int m_lane = lane & 15;
  const int half   = lane >> 4;
  const float scale = 0.17677669529663687f;  // 1/sqrt(32)

  // --- stage q tile, relK (padded), relV^T (padded), zero bins ---
  for (int i = lane; i < 16 * 32; i += 32) {
    int r = i >> 5, c = i & 31;
    qtile[r][c] = (_Float16)qg[(size_t)(row0 + r) * HIDDEN + h * DHEAD + c];
  }
  for (int i = lane; i < RVP * 32; i += 32) {
    int r = i >> 5, c = i & 31;
    relk_h[r][c] = (r < RVOC) ? (_Float16)relK[r * DHEAD + c] : (_Float16)0.0f;
  }
  for (int i = lane; i < 32 * RVP; i += 32) {
    int d = i >> 6, r = i & 63;
    relvt[d][r] = (r < RVOC) ? (_Float16)relV[r * DHEAD + d] : (_Float16)0.0f;
  }
  for (int i = lane; i < 16 * RVP; i += 32) prbin[i >> 6][i & 63] = 0.0f;
  __syncthreads();

  v16h aq = load_frag_h(&qtile[0][0], m_lane, half, 40);

  // --- qrel[q][r] = q . relK[r]  (16x32x64, 4 WMMAs) ---
#pragma unroll
  for (int rt = 0; rt < 4; ++rt) {
    v8f acc = {};
    v16h bf = load_frag_h(&relk_h[rt * 16][0], m_lane, half, 40);
    acc = wmma_f16(aq, bf, acc);
#pragma unroll
    for (int r = 0; r < 8; ++r) qrel[r + 8 * half][rt * 16 + m_lane] = acc[r];
  }
  __syncthreads();

  // --- scores: loop over 32 key tiles of 16 ---
  for (int kt = 0; kt < 32; ++kt) {
    for (int i = lane; i < 16 * 32; i += 32) {
      int r = i >> 5, c = i & 31;
      ktile[r][c] = (_Float16)kg[(size_t)(b * SEQ + kt * 16 + r) * HIDDEN + h * DHEAD + c];
    }
    __syncthreads();
    v16h bf = load_frag_h(&ktile[0][0], m_lane, half, 40);
    v8f acc = {};
    acc = wmma_f16(aq, bf, acc);
    const int kcol = kt * 16 + m_lane;
#pragma unroll
    for (int r = 0; r < 8; ++r) {
      int m = r + 8 * half;
      int ri = rel[(size_t)(row0 + m) * SEQ + kcol];
      float val = (acc[r] + qrel[m][ri]) * scale;
      if (pmask[b * SEQ + kcol]) val = -3.0e38f;
      scores[m][kcol] = val;
      relid[m][kcol] = (unsigned char)ri;
    }
    __syncthreads();
  }

  // --- softmax per row + bincount of probs over rel ids ---
  for (int m = 0; m < 16; ++m) {
    float vmax = -3.0e38f;
    for (int c = lane; c < SEQ; c += 32) vmax = fmaxf(vmax, scores[m][c]);
#pragma unroll
    for (int off = 16; off > 0; off >>= 1)
      vmax = fmaxf(vmax, __shfl_xor(vmax, off, 32));
    float vsum = 0.0f;
    for (int c = lane; c < SEQ; c += 32) {
      float e = __expf(scores[m][c] - vmax);
      scores[m][c] = e;
      vsum += e;
    }
#pragma unroll
    for (int off = 16; off > 0; off >>= 1) vsum += __shfl_xor(vsum, off, 32);
    float inv = 1.0f / vsum;
    for (int c = lane; c < SEQ; c += 32) {
      float p = scores[m][c] * inv;
      scores[m][c] = p;
      atomicAdd(&prbin[m][relid[m][c]], p);   // ds_add_f32
    }
  }
  __syncthreads();

  // --- ctx = probs @ V  (16 chunks of 32 keys, 2 d-tiles) ---
  v8f acc0 = {}, acc1 = {};
  for (int kc = 0; kc < 16; ++kc) {
    for (int i = lane; i < 32 * 32; i += 32) {
      int d = i >> 5, kk = i & 31;
      vtile[d][kk] = (_Float16)vg[(size_t)(b * SEQ + kc * 32 + kk) * HIDDEN + h * DHEAD + d];
    }
    __syncthreads();
    v16h ap = load_frag_f32(&scores[0][kc * 32], m_lane, half, SEQ);
    v16h b0 = load_frag_h(&vtile[0][0],  m_lane, half, 40);
    v16h b1 = load_frag_h(&vtile[16][0], m_lane, half, 40);
    acc0 = wmma_f16(ap, b0, acc0);
    acc1 = wmma_f16(ap, b1, acc1);
    __syncthreads();
  }

  // --- ctx += bincount(probs) @ relV  (2 r-chunks x 2 d-tiles) ---
#pragma unroll
  for (int rc = 0; rc < 2; ++rc) {
    v16h ap = load_frag_f32(&prbin[0][rc * 32], m_lane, half, RVP);
    v16h b0 = load_frag_h(&relvt[0][rc * 32],  m_lane, half, 72);
    v16h b1 = load_frag_h(&relvt[16][rc * 32], m_lane, half, 72);
    acc0 = wmma_f16(ap, b0, acc0);
    acc1 = wmma_f16(ap, b1, acc1);
  }

#pragma unroll
  for (int r = 0; r < 8; ++r) {
    int m = r + 8 * half;
    ctx[(size_t)(row0 + m) * HIDDEN + h * DHEAD + m_lane]      = acc0[r];
    ctx[(size_t)(row0 + m) * HIDDEN + h * DHEAD + 16 + m_lane] = acc1[r];
  }
}

// ---------------- residual + LayerNorm ----------------
__global__ __launch_bounds__(256) void ln_kernel(
    const float* __restrict__ x, const float* __restrict__ res,
    const float* __restrict__ g, const float* __restrict__ bb,
    float* __restrict__ out) {
  __shared__ float red[256];
  const int row = blockIdx.x, t = threadIdx.x;
  float val = x[(size_t)row * HIDDEN + t] + res[(size_t)row * HIDDEN + t];
  red[t] = val;
  __syncthreads();
  for (int s = 128; s > 0; s >>= 1) {
    if (t < s) red[t] += red[t + s];
    __syncthreads();
  }
  float mu = red[0] / HIDDEN;
  __syncthreads();
  float d = val - mu;
  red[t] = d * d;
  __syncthreads();
  for (int s = 128; s > 0; s >>= 1) {
    if (t < s) red[t] += red[t + s];
    __syncthreads();
  }
  float var = red[0] / HIDDEN;
  out[(size_t)row * HIDDEN + t] = d * rsqrtf(var + 1e-5f) * g[t] + bb[t];
}

// ---------------- host orchestration ----------------
extern "C" void kernel_launch(void* const* d_in, const int* in_sizes, int n_in,
                              void* d_out, int out_size, void* d_ws,
                              size_t ws_size, hipStream_t stream) {
  (void)in_sizes; (void)n_in; (void)out_size; (void)ws_size;
  const float* x     = (const float*)d_in[0];
  const float* Wq    = (const float*)d_in[1];
  const float* Wk    = (const float*)d_in[2];
  const float* Wv    = (const float*)d_in[3];
  const float* Wd    = (const float*)d_in[4];
  const float* bd    = (const float*)d_in[5];
  const float* relK  = (const float*)d_in[6];
  const float* relV  = (const float*)d_in[7];
  const float* W1    = (const float*)d_in[8];
  const float* b1    = (const float*)d_in[9];
  const float* W2    = (const float*)d_in[10];
  const float* b2    = (const float*)d_in[11];
  const float* ln1g  = (const float*)d_in[12];
  const float* ln1b  = (const float*)d_in[13];
  const float* ln2g  = (const float*)d_in[14];
  const float* ln2b  = (const float*)d_in[15];
  const int*   rel   = (const int*)d_in[16];
  const unsigned char* pmask = (const unsigned char*)d_in[17];

  const size_t act = (size_t)MROWS * HIDDEN;  // 524288 floats
  float* ws   = (float*)d_ws;
  float* xcur = ws;
  float* qb   = xcur + act;
  float* kb   = qb + act;
  float* vb   = kb + act;
  float* ctx  = vb + act;
  float* tmp  = ctx + act;                    // attn_out / ffn2 output
  float* ffb  = tmp + act;                    // MROWS * FFDIM

  // f16 N-major weight shadow copies (converted once per launch)
  const size_t sq = (size_t)HIDDEN * HIDDEN;      // 65536
  const size_t sf = (size_t)HIDDEN * FFDIM;       // 262144
  _Float16* h16 = (_Float16*)(ffb + (size_t)MROWS * FFDIM);
  _Float16* whq = h16;                    // [NLAYER][HIDDEN][HIDDEN]
  _Float16* whk = whq + NLAYER * sq;
  _Float16* whv = whk + NLAYER * sq;
  _Float16* whd = whv + NLAYER * sq;
  _Float16* wh1 = whd + NLAYER * sq;      // [NLAYER][FFDIM][HIDDEN]
  _Float16* wh2 = wh1 + NLAYER * sf;      // [NLAYER][HIDDEN][FFDIM]

  (void)hipMemcpyAsync(xcur, x, act * sizeof(float), hipMemcpyDeviceToDevice,
                       stream);

  const dim3 blkC(256);
  const dim3 grdCq((sq + 255) / 256), grdCf((sf + 255) / 256);
  for (int l = 0; l < NLAYER; ++l) {
    cvt_transpose_kernel<<<grdCq, blkC, 0, stream>>>(Wq + l * sq, whq + l * sq,
                                                     HIDDEN, HIDDEN);
    cvt_transpose_kernel<<<grdCq, blkC, 0, stream>>>(Wk + l * sq, whk + l * sq,
                                                     HIDDEN, HIDDEN);
    cvt_transpose_kernel<<<grdCq, blkC, 0, stream>>>(Wv + l * sq, whv + l * sq,
                                                     HIDDEN, HIDDEN);
    cvt_transpose_kernel<<<grdCq, blkC, 0, stream>>>(Wd + l * sq, whd + l * sq,
                                                     HIDDEN, HIDDEN);
    cvt_transpose_kernel<<<grdCf, blkC, 0, stream>>>(W1 + l * sf, wh1 + l * sf,
                                                     HIDDEN, FFDIM);
    cvt_transpose_kernel<<<grdCf, blkC, 0, stream>>>(W2 + l * sf, wh2 + l * sf,
                                                     FFDIM, HIDDEN);
  }

  const dim3 blkG(128), blkA(32), blkL(256);
  const dim3 grdProj(HIDDEN / 64, MROWS / 64);   // (4, 32)
  const dim3 grdFF1(FFDIM / 64,  MROWS / 64);    // (16, 32)
  const dim3 grdFF2(HIDDEN / 64, MROWS / 64);
  const dim3 grdAttn(BATCH * NHEAD * (SEQ / 16)); // 1024
  const dim3 grdLN(MROWS);

  for (int l = 0; l < NLAYER; ++l) {
    const float* bd_l = bd + (size_t)l * HIDDEN;
    const float* rk_l = relK + (size_t)l * RVOC * DHEAD;
    const float* rv_l = relV + (size_t)l * RVOC * DHEAD;
    const float* b1_l = b1 + (size_t)l * FFDIM;
    const float* b2_l = b2 + (size_t)l * HIDDEN;

    gemm_kernel<<<grdProj, blkG, 0, stream>>>(xcur, whq + l * sq, nullptr, qb,
                                              MROWS, HIDDEN, HIDDEN, 0);
    gemm_kernel<<<grdProj, blkG, 0, stream>>>(xcur, whk + l * sq, nullptr, kb,
                                              MROWS, HIDDEN, HIDDEN, 0);
    gemm_kernel<<<grdProj, blkG, 0, stream>>>(xcur, whv + l * sq, nullptr, vb,
                                              MROWS, HIDDEN, HIDDEN, 0);

    attn_kernel<<<grdAttn, blkA, 0, stream>>>(qb, kb, vb, rk_l, rv_l, rel,
                                              pmask, ctx);

    gemm_kernel<<<grdProj, blkG, 0, stream>>>(ctx, whd + l * sq, bd_l, tmp,
                                              MROWS, HIDDEN, HIDDEN, 0);
    ln_kernel<<<grdLN, blkL, 0, stream>>>(xcur, tmp,
                                          ln1g + (size_t)l * HIDDEN,
                                          ln1b + (size_t)l * HIDDEN, xcur);

    gemm_kernel<<<grdFF1, blkG, 0, stream>>>(xcur, wh1 + l * sf, b1_l, ffb,
                                             MROWS, HIDDEN, FFDIM, 1);
    gemm_kernel<<<grdFF2, blkG, 0, stream>>>(ffb, wh2 + l * sf, b2_l, tmp,
                                             MROWS, FFDIM, HIDDEN, 0);

    float* lnout = (l == NLAYER - 1) ? (float*)d_out : xcur;
    ln_kernel<<<grdLN, blkL, 0, stream>>>(xcur, tmp,
                                          ln2g + (size_t)l * HIDDEN,
                                          ln2b + (size_t)l * HIDDEN, lnout);
  }
}
